// MGNNE_38671885533491
// MI455X (gfx1250) — 8-bit matrix operand order repaired and checked by reading, compiled, not run on hardware
//
#include <hip/hip_runtime.h>
#include <math.h>

#define NN 64
#define SL 32
#define LL 128

typedef __attribute__((ext_vector_type(8))) int v8i;

struct Params {
    const int* A;
    const float* ew1[3]; const float* eb1[3];
    const float* ew2[3]; const float* eb2[3];
    const float* ew3[3]; const float* eb3[3];
    const float* ww1; const float* wb1;
    const float* ww2; const float* wb2;
    const float* ww3; const float* wb3;
    float* G;
    float* out;
};

__global__ __launch_bounds__(128) void g_tables_kernel(Params p) {
    __shared__ float Gs[960];
    __shared__ float H1[81 * SL];
    __shared__ float H2[81 * SL];
    const int offs[5]   = {0, 0, 24, 96, 312};
    const int counts[5] = {0, 3, 9, 27, 81};
    int tid = threadIdx.x;

    for (int lvl = 1; lvl <= 4; ++lvl) {
        int E = counts[lvl];
        for (int idx = tid; idx < E * SL; idx += blockDim.x) {
            int e = idx / SL, t = idx % SL;
            int r = e / 3, q = e % 3;
            float acc = p.eb1[q][t];
            #pragma unroll
            for (int i = 0; i < 16; ++i) {
                float xi = (i < 8) ? ((lvl == 1) ? 0.0f : Gs[offs[lvl - 1] + r * 8 + i])
                                   : 1.0f;
                acc += xi * p.ew1[q][i * SL + t];
            }
            H1[idx] = tanhf(acc);
        }
        __syncthreads();
        for (int idx = tid; idx < E * SL; idx += blockDim.x) {
            int e = idx / SL, t = idx % SL, q = e % 3;
            float acc = p.eb2[q][t];
            for (int i = 0; i < SL; ++i) acc += H1[e * SL + i] * p.ew2[q][i * SL + t];
            H2[idx] = tanhf(acc);
        }
        __syncthreads();
        for (int idx = tid; idx < E * 8; idx += blockDim.x) {
            int e = idx / 8, t = idx % 8, q = e % 3;
            float acc = p.eb3[q][t];
            for (int i = 0; i < SL; ++i) acc += H2[e * SL + i] * p.ew3[q][i * 8 + t];
            Gs[offs[lvl] + idx] = acc;
        }
        __syncthreads();
    }
    for (int i = tid; i < 960; i += blockDim.x) p.G[i] = Gs[i];
}

__global__ __launch_bounds__(256) void node_kernel(Params p) {
    const int n    = blockIdx.x;
    const int tid  = threadIdx.x;
    const int lane = tid & 31;
    const int wv   = tid >> 5;

    __shared__ __align__(16) signed char t8[3][NN][NN];
    __shared__ int   Q[NN * NN];
    __shared__ int   deg[3][NN];
    __shared__ int   mv[3][3][NN];
    __shared__ int   path3[NN][27];
    __shared__ int   red[256][6];
    __shared__ int   Arow[NN];
    __shared__ float cnt[120];
    __shared__ float y[40];
    __shared__ float h1[LL], h2[LL];

    for (int idx = tid; idx < NN * NN; idx += 256) {
        int i = idx >> 6, j = idx & 63;
        int av = p.A[idx];
        int del = (i == n) | (j == n);
        t8[0][i][j] = (signed char)((av == 1 && !del) ? 1 : 0);
        t8[1][i][j] = (signed char)((av == 2 && !del) ? 1 : 0);
        t8[2][i][j] = (signed char)((av == 3 && !del) ? 1 : 0);
    }
    if (tid < NN) Arow[tid] = p.A[n * NN + tid];
    __syncthreads();

    if (tid < 192) {
        int a = tid / 64, i = tid & 63;
        int s = 0;
        for (int j = 0; j < NN; ++j) s += (int)t8[a][i][j];
        deg[a][i] = s;
    }
    __syncthreads();

    for (int idx = tid; idx < 576; idx += 256) {
        int x = idx / 192, rem = idx % 192, a = rem / 64, i = rem & 63;
        int s = 0;
        for (int j = 0; j < NN; ++j) s += (int)t8[x][i][j] * deg[a][j];
        mv[x][a][i] = s;
    }
    __syncthreads();

    const int half = lane >> 4, l15 = lane & 15;

    for (int cb = 0; cb < 9; ++cb) {
        int c = cb / 3, b = cb % 3;
        #pragma unroll
        for (int rep = 0; rep < 2; ++rep) {
            int t  = wv + rep * 8;
            int ti = t >> 2, tj = t & 3;
            const unsigned int* arow = (const unsigned int*)&t8[c][ti * 16 + l15][0];
            const unsigned int* brow = (const unsigned int*)&t8[b][tj * 16 + l15][0];
            v8i va, vb;
            #pragma unroll
            for (int v = 0; v < 8; ++v) {
                va[v] = (int)arow[(v >> 1) * 4 + (v & 1) + half * 2];
                vb[v] = (int)brow[(v >> 1) * 4 + (v & 1) + half * 2];
            }
            v8i vc = {0, 0, 0, 0, 0, 0, 0, 0};
            v8i vd = __builtin_amdgcn_wmma_i32_16x16x64_iu8(false, va, false, vb, vc,
                                                            false, false);
            #pragma unroll
            for (int r = 0; r < 8; ++r)
                Q[(ti * 16 + r + 8 * half) * NN + tj * 16 + l15] = vd[r];
        }
        __syncthreads();

        {
            int i  = tid >> 2;
            int j0 = (tid & 3) * 16;
            int s30 = 0, s31 = 0, s32 = 0, d30 = 0, d31 = 0, d32 = 0;
            #pragma unroll
            for (int jj = 0; jj < 16; ++jj) {
                int j = j0 + jj;
                int q = Q[i * NN + j];
                s30 += q * deg[0][j]; s31 += q * deg[1][j]; s32 += q * deg[2][j];
                d30 += q * (int)t8[0][i][j];
                d31 += q * (int)t8[1][i][j];
                d32 += q * (int)t8[2][i][j];
            }
            red[tid][0] = s30; red[tid][1] = s31; red[tid][2] = s32;
            red[tid][3] = d30; red[tid][4] = d31; red[tid][5] = d32;
        }
        __syncthreads();

        if (tid < NN) {
            int i = tid;
            int s3[3], d3[3];
            #pragma unroll
            for (int a = 0; a < 3; ++a) {
                s3[a] = red[i * 4 + 0][a] + red[i * 4 + 1][a] +
                        red[i * 4 + 2][a] + red[i * 4 + 3][a];
                d3[a] = red[i * 4 + 0][3 + a] + red[i * 4 + 1][3 + a] +
                        red[i * 4 + 2][3 + a] + red[i * 4 + 3][3 + a];
            }
            #pragma unroll
            for (int a = 0; a < 3; ++a) {
                int v = s3[a] - d3[a];
                if (b == c) v -= deg[c][i] * deg[a][i];
                if (a == b) v -= mv[c][a][i];
                if (a == b && b == c) v += deg[a][i];
                path3[i][a * 9 + b * 3 + c] = v;
            }
        }
        __syncthreads();
    }

    if (tid < 81) {
        int d = tid % 3, c = (tid / 3) % 3, b = (tid / 9) % 3, a = tid / 27;
        int s = 0;
        for (int i = 0; i < NN; ++i)
            if (Arow[i] == d + 1) s += path3[i][a * 9 + b * 3 + c];
        cnt[tid] = (float)s;
    } else if (tid < 108) {
        int u = tid - 81;
        int c = u % 3, b = (u / 3) % 3, a = u / 9;
        int s = 0;
        for (int i = 0; i < NN; ++i)
            if (Arow[i] == c + 1) s += mv[b][a][i] - ((a == b) ? deg[a][i] : 0);
        cnt[tid] = (float)s;
    } else if (tid < 117) {
        int u = tid - 108;
        int b = u % 3, a = u / 3;
        int s = 0;
        for (int i = 0; i < NN; ++i)
            if (Arow[i] == b + 1) s += deg[a][i];
        cnt[tid] = (float)s;
    } else if (tid < 120) {
        int a = tid - 117;
        int s = 0;
        for (int j = 0; j < NN; ++j)
            if (Arow[j] == a + 1) s++;
        cnt[tid] = (float)s;
    }
    __syncthreads();

    if (tid < 8) {
        int f = tid;
        const float* G1 = p.G;
        const float* G2 = p.G + 24;
        const float* G3 = p.G + 96;
        const float* G4 = p.G + 312;
        float v0 = 0, v1 = 0, v2 = 0, v3 = 0;
        for (int a = 0; a < 3; ++a)  v0 += cnt[117 + a] * G1[a * 8 + f];
        for (int u = 0; u < 9; ++u)  v1 += cnt[108 + u] * G2[u * 8 + f];
        for (int u = 0; u < 27; ++u) v2 += cnt[81 + u] * G3[u * 8 + f];
        for (int u = 0; u < 81; ++u) v3 += cnt[u] * G4[u * 8 + f];
        y[f * 4 + 0] = v0; y[f * 4 + 1] = v1; y[f * 4 + 2] = v2; y[f * 4 + 3] = v3;
    }
    if (tid >= 32 && tid < 40) y[tid] = 1.0f;
    __syncthreads();

    if (tid < LL) {
        float acc = p.wb1[tid];
        for (int i = 0; i < 40; ++i) acc += y[i] * p.ww1[i * LL + tid];
        h1[tid] = tanhf(acc);
    }
    __syncthreads();
    if (tid < LL) {
        float acc = p.wb2[tid];
        for (int i = 0; i < LL; ++i) acc += h1[i] * p.ww2[i * LL + tid];
        h2[tid] = tanhf(acc);
    }
    __syncthreads();
    if (tid == 0) {
        float acc = p.wb3[0];
        for (int i = 0; i < LL; ++i) acc += h2[i] * p.ww3[i];
        p.out[n] = acc;
    }
}

extern "C" void kernel_launch(void* const* d_in, const int* in_sizes, int n_in,
                              void* d_out, int out_size, void* d_ws, size_t ws_size,
                              hipStream_t stream) {
    (void)in_sizes; (void)n_in; (void)out_size; (void)ws_size;
    Params p;
    p.A = (const int*)d_in[0];
    for (int q = 0; q < 3; ++q) {
        p.ew1[q] = (const float*)d_in[1 + 6 * q + 0];
        p.eb1[q] = (const float*)d_in[1 + 6 * q + 1];
        p.ew2[q] = (const float*)d_in[1 + 6 * q + 2];
        p.eb2[q] = (const float*)d_in[1 + 6 * q + 3];
        p.ew3[q] = (const float*)d_in[1 + 6 * q + 4];
        p.eb3[q] = (const float*)d_in[1 + 6 * q + 5];
    }
    p.ww1 = (const float*)d_in[19]; p.wb1 = (const float*)d_in[20];
    p.ww2 = (const float*)d_in[21]; p.wb2 = (const float*)d_in[22];
    p.ww3 = (const float*)d_in[23]; p.wb3 = (const float*)d_in[24];
    p.G   = (float*)d_ws;
    p.out = (float*)d_out;

    hipLaunchKernelGGL(g_tables_kernel, dim3(1), dim3(128), 0, stream, p);
    hipLaunchKernelGGL(node_kernel, dim3(NN), dim3(256), 0, stream, p);
}
